// SimpleRegressionOutput_9758165697299
// MI455X (gfx1250) — compile-verified, matrix operand order checked by reading, not run on hardware
//
#include <hip/hip_runtime.h>
#include <math.h>

// ---------------------------------------------------------------------------
// Problem constants (from the reference)
// ---------------------------------------------------------------------------
#define BATCH   512
#define NSEQ    100
#define HDIM    768            // K dimension of big GEMM
#define IDIM    1536           // N dimension of big GEMM
#define MROWS   (BATCH * NSEQ) // 51200 rows
#define LN_EPS  1e-5f

// WMMA tiling
#define TILE_M     16
#define TILE_N     16
#define TILE_K     32
#define KSTEPS     (HDIM / TILE_K)        // 24
#define NTILES     (IDIM / TILE_N)        // 96
#define WAVES_PER_WG 8
#define TILES_PER_WAVE (NTILES / WAVES_PER_WG)  // 12
#define COLS_PER_WAVE  (TILES_PER_WAVE * TILE_N) // 192

typedef __attribute__((ext_vector_type(16))) __bf16 v16bf;
typedef __attribute__((ext_vector_type(8)))  float  v8f;

union BFrag { v16bf v; unsigned short u[16]; };

// float -> bf16 (round-to-nearest-even), bit-exact and independent of __bf16
// conversion codegen.
__device__ __forceinline__ unsigned short f2bf(float f) {
    unsigned int u = __float_as_uint(f);
    u += 0x7FFFu + ((u >> 16) & 1u);
    return (unsigned short)(u >> 16);
}

__device__ __forceinline__ float gelu_erf(float x) {
    // exact GELU: 0.5*x*(1+erf(x/sqrt(2)))
    return 0.5f * x * (1.0f + erff(x * 0.70710678118654752f));
}

// ---------------------------------------------------------------------------
// Kernel 1: pack W1 [H,I] fp32 row-major -> bf16 B-fragments in ws.
// Fragment layout (16-bit B matrix 32x16, per CDNA5 ISA 7.12.2):
//   lane n (0..15) holds column n, lanes 16..31 hold column (lane-16);
//   element e of the lane's 16 bf16 values maps to
//   K = e + (e>=8 ? 8 : 0) + 8*(lane>>4).
// Tiles stored as [kt][nt][lane][e] so one K-step of a workgroup is a
// contiguous 96 KB stream (L2-resident).
// ---------------------------------------------------------------------------
__global__ __launch_bounds__(256) void pack_w1_kernel(
    const float* __restrict__ W1, unsigned short* __restrict__ Wb)
{
    unsigned int idx = blockIdx.x * 256u + threadIdx.x;   // dest element index
    // total = 24*96*32*16 = 1,179,648 elements, grid sized exactly
    unsigned int e    = idx & 15u;
    unsigned int lane = (idx >> 4) & 31u;
    unsigned int tile = idx >> 9;            // kt*96 + nt
    unsigned int kt   = tile / NTILES;
    unsigned int nt   = tile % NTILES;

    unsigned int n = nt * TILE_N + (lane & 15u);
    unsigned int k = kt * TILE_K + e + ((e >= 8u) ? 8u : 0u) + 8u * (lane >> 4);

    Wb[idx] = f2bf(W1[(size_t)k * IDIM + n]);
}

// ---------------------------------------------------------------------------
// Kernel 2: fused  GEMM(bf16 WMMA, f32 acc) + bias + GELU + LayerNorm +
//           GEMV(W2) + sigmoid.  One WG = 16 rows x 1536 cols, 8 waves.
// ---------------------------------------------------------------------------
__global__ __launch_bounds__(256) void fused_mlp_kernel(
    const float* __restrict__ v_emb,
    const unsigned short* __restrict__ Wb,   // packed bf16 W1 fragments
    const float* __restrict__ b1,
    const float* __restrict__ gamma,
    const float* __restrict__ beta,
    const float* __restrict__ W2,
    const float* __restrict__ b2,
    float* __restrict__ scores)              // [MROWS]
{
    const int wave = threadIdx.x >> 5;       // 0..7
    const int lane = threadIdx.x & 31;
    const int m    = lane & 15;              // row-within-tile / col-within-tile
    const int g    = lane >> 4;              // lane group
    const int rowbase = blockIdx.x * TILE_M;
    const int colbase = wave * COLS_PER_WAVE;

    // ---- accumulators: 12 tiles of 16x16 f32 (8 VGPRs each) ----
    v8f acc[TILES_PER_WAVE];
    #pragma unroll
    for (int t = 0; t < TILES_PER_WAVE; ++t) acc[t] = (v8f)0.0f;

    // A: lane holds row (rowbase+m); K runs [8g..8g+7] and [16+8g..16+8g+7]
    const float* arow = v_emb + (size_t)(rowbase + m) * HDIM + 8 * g;
    // B: this wave's 12 tiles for K-step kt start here
    const unsigned short* bbase = Wb + ((size_t)wave * TILES_PER_WAVE * 512) + lane * 16;

    for (int kt = 0; kt < KSTEPS; ++kt) {
        const unsigned short* bk = bbase + (size_t)kt * (NTILES * 512);
        if (kt + 1 < KSTEPS) {
            // pull next K-step's B strip toward L0/L2 (global_prefetch_b8)
            __builtin_prefetch(bbase + (size_t)(kt + 1) * (NTILES * 512), 0, 1);
        }

        // ---- build A fragment (16 fp32 -> 16 bf16, ISA element order) ----
        const float4* ap = reinterpret_cast<const float4*>(arow + kt * TILE_K);
        float4 q0 = ap[0];            // K = 8g+0..3
        float4 q1 = ap[1];            // K = 8g+4..7
        const float4* ap2 = reinterpret_cast<const float4*>(arow + kt * TILE_K + 16);
        float4 q2 = ap2[0];           // K = 16+8g+0..3
        float4 q3 = ap2[1];           // K = 16+8g+4..7

        BFrag af;
        af.u[0]  = f2bf(q0.x); af.u[1]  = f2bf(q0.y);
        af.u[2]  = f2bf(q0.z); af.u[3]  = f2bf(q0.w);
        af.u[4]  = f2bf(q1.x); af.u[5]  = f2bf(q1.y);
        af.u[6]  = f2bf(q1.z); af.u[7]  = f2bf(q1.w);
        af.u[8]  = f2bf(q2.x); af.u[9]  = f2bf(q2.y);
        af.u[10] = f2bf(q2.z); af.u[11] = f2bf(q2.w);
        af.u[12] = f2bf(q3.x); af.u[13] = f2bf(q3.y);
        af.u[14] = f2bf(q3.z); af.u[15] = f2bf(q3.w);

        #pragma unroll
        for (int t = 0; t < TILES_PER_WAVE; ++t) {
            v16bf bf = *reinterpret_cast<const v16bf*>(bk + (size_t)t * 512);
            acc[t] = __builtin_amdgcn_wmma_f32_16x16x32_bf16(
                /*neg_a=*/false, af.v, /*neg_b=*/false, bf,
                /*c_mod=*/(short)0, acc[t],
                /*reuse_a=*/false, /*reuse_b=*/false);
        }
    }

    // ---- epilogue: per-column constants ----
    float b1v[TILES_PER_WAVE], gv[TILES_PER_WAVE], bev[TILES_PER_WAVE], w2v[TILES_PER_WAVE];
    #pragma unroll
    for (int t = 0; t < TILES_PER_WAVE; ++t) {
        int col = colbase + t * TILE_N + m;
        b1v[t] = b1[col];
        gv[t]  = gamma[col];
        bev[t] = beta[col];
        w2v[t] = W2[col];
    }

    // C/D layout: VGPR r, lanes 0-15 -> row r, lanes 16-31 -> row r+8
    // => this lane's register r holds row (r + 8g), column (colbase + 16t + m).

    // ---- bias + exact GELU, accumulate row sum / sumsq ----
    float s1[8], s2[8];
    #pragma unroll
    for (int r = 0; r < 8; ++r) { s1[r] = 0.0f; s2[r] = 0.0f; }
    #pragma unroll
    for (int t = 0; t < TILES_PER_WAVE; ++t) {
        #pragma unroll
        for (int r = 0; r < 8; ++r) {
            float x = acc[t][r] + b1v[t];
            x = gelu_erf(x);
            acc[t][r] = x;
            s1[r] += x;
            s2[r] += x * x;
        }
    }
    // reduce across the 16 lanes that share a row (xor over low 4 lane bits)
    #pragma unroll
    for (int mask = 1; mask <= 8; mask <<= 1) {
        #pragma unroll
        for (int r = 0; r < 8; ++r) {
            s1[r] += __shfl_xor(s1[r], mask, 32);
            s2[r] += __shfl_xor(s2[r], mask, 32);
        }
    }

    __shared__ float red1[TILE_M], red2[TILE_M], muS[TILE_M], rsS[TILE_M], ssc[TILE_M];
    if (threadIdx.x < TILE_M) {
        red1[threadIdx.x] = 0.0f; red2[threadIdx.x] = 0.0f; ssc[threadIdx.x] = 0.0f;
    }
    __syncthreads();
    if (m == 0) {   // lanes 0 and 16 carry the wave's row totals
        #pragma unroll
        for (int r = 0; r < 8; ++r) {
            atomicAdd(&red1[r + 8 * g], s1[r]);   // ds_add_f32
            atomicAdd(&red2[r + 8 * g], s2[r]);
        }
    }
    __syncthreads();
    if (threadIdx.x < TILE_M) {
        float mu  = red1[threadIdx.x] * (1.0f / (float)IDIM);
        float var = red2[threadIdx.x] * (1.0f / (float)IDIM) - mu * mu;
        muS[threadIdx.x] = mu;
        rsS[threadIdx.x] = rsqrtf(var + LN_EPS);
    }
    __syncthreads();

    float mur[8], rsr[8];
    #pragma unroll
    for (int r = 0; r < 8; ++r) { mur[r] = muS[r + 8 * g]; rsr[r] = rsS[r + 8 * g]; }

    // ---- LayerNorm + dot with W2 ----
    float sc[8];
    #pragma unroll
    for (int r = 0; r < 8; ++r) sc[r] = 0.0f;
    #pragma unroll
    for (int t = 0; t < TILES_PER_WAVE; ++t) {
        #pragma unroll
        for (int r = 0; r < 8; ++r) {
            float xn = (acc[t][r] - mur[r]) * rsr[r] * gv[t] + bev[t];
            sc[r] += xn * w2v[t];
        }
    }
    #pragma unroll
    for (int mask = 1; mask <= 8; mask <<= 1) {
        #pragma unroll
        for (int r = 0; r < 8; ++r) sc[r] += __shfl_xor(sc[r], mask, 32);
    }
    if (m == 0) {
        #pragma unroll
        for (int r = 0; r < 8; ++r) atomicAdd(&ssc[r + 8 * g], sc[r]);
    }
    __syncthreads();
    if (threadIdx.x < TILE_M) {
        float s = ssc[threadIdx.x] + b2[0];
        scores[rowbase + threadIdx.x] = 1.0f / (1.0f + expf(-s));   // sigmoid
    }
}

// ---------------------------------------------------------------------------
// Kernel 3: pred = sum_n scores[b,n];  logits = one_hot(clip(rint(pred),0,15))
// ---------------------------------------------------------------------------
__global__ __launch_bounds__(128) void finalize_kernel(
    const float* __restrict__ scores,   // [B, N]
    float* __restrict__ pred,           // [B]
    float* __restrict__ logits)         // [B, 16]
{
    const int b   = blockIdx.x;
    const int tid = threadIdx.x;

    __shared__ float tot;
    if (tid == 0) tot = 0.0f;
    __syncthreads();

    float v = (tid < NSEQ) ? scores[(size_t)b * NSEQ + tid] : 0.0f;
    #pragma unroll
    for (int mask = 16; mask >= 1; mask >>= 1) v += __shfl_xor(v, mask, 32);
    if ((tid & 31) == 0) atomicAdd(&tot, v);
    __syncthreads();

    float P = tot;
    if (tid == 0) pred[b] = P;
    if (tid < 16) {
        int aid = (int)rintf(P);                 // round half-to-even (RNE)
        aid = aid < 0 ? 0 : (aid > 15 ? 15 : aid);
        logits[(size_t)b * 16 + tid] = (tid == aid) ? 1.0f : 0.0f;
    }
}

// ---------------------------------------------------------------------------
// Launch
// ---------------------------------------------------------------------------
extern "C" void kernel_launch(void* const* d_in, const int* in_sizes, int n_in,
                              void* d_out, int out_size, void* d_ws, size_t ws_size,
                              hipStream_t stream) {
    (void)in_sizes; (void)n_in; (void)out_size; (void)ws_size;

    const float* v_emb = (const float*)d_in[0];   // [B,N,H]
    const float* W1    = (const float*)d_in[1];   // [H,I]
    const float* b1    = (const float*)d_in[2];   // [I]
    const float* gamma = (const float*)d_in[3];   // [I]
    const float* beta  = (const float*)d_in[4];   // [I]
    const float* W2    = (const float*)d_in[5];   // [I,1]
    const float* b2    = (const float*)d_in[6];   // [1]

    float* out      = (float*)d_out;
    float* scores   = out;                         // [B*N]
    float* pred     = out + (size_t)MROWS;         // [B]
    float* logits   = pred + BATCH;                // [B,16]

    unsigned short* Wb = (unsigned short*)d_ws;    // 2.25 MB packed bf16 W1

    // 1) pack W1 into WMMA B-fragment order (bf16)
    const int packElems  = KSTEPS * NTILES * 512;  // 1,179,648
    pack_w1_kernel<<<packElems / 256, 256, 0, stream>>>(W1, Wb);

    // 2) fused GEMM + GELU + LayerNorm + GEMV + sigmoid
    fused_mlp_kernel<<<MROWS / TILE_M, 256, 0, stream>>>(
        v_emb, Wb, b1, gamma, beta, W2, b2, scores);

    // 3) per-batch reduction + one-hot logits
    finalize_kernel<<<BATCH, 128, 0, stream>>>(scores, pred, logits);
}
